// GRUNetv3_extra_attn_48284022342319
// MI455X (gfx1250) — matrix operand order repaired and checked by reading, not compiled, not run on hardware
//
#include <hip/hip_runtime.h>
#include <hip/hip_bf16.h>
#include <math.h>

typedef __bf16 bf16;
typedef __attribute__((ext_vector_type(16))) __bf16 v16bf;
typedef __attribute__((ext_vector_type(8)))  __bf16 v8bf;
typedef __attribute__((ext_vector_type(8)))  float   v8f;
typedef __attribute__((ext_vector_type(4)))  unsigned int u32x4;
typedef __attribute__((ext_vector_type(8)))  int i32x8;
typedef __attribute__((ext_vector_type(4)))  int i32x4;

#define S_LEN 256
#define BATCH 512
#define IDIM  64
#define HDIM  128
#define DDIM  32
#define G3H   384
#define NTOK  (S_LEN * BATCH)   // 131072

// ---------------------------------------------------------------------------
// WMMA helpers (CDNA5 v_wmma_f32_16x16x32_bf16, wave32)
// ---------------------------------------------------------------------------
__device__ __forceinline__ v8f wmma_bf(v16bf a, v16bf b, v8f c) {
  return __builtin_amdgcn_wmma_f32_16x16x32_bf16(false, a, false, b, (short)0, c,
                                                 false, false);
}

// A fragment: 16x32 bf16, row-major source with row stride `stride` (elements).
__device__ __forceinline__ v16bf load_a16x32(const bf16* base, int stride, int k0) {
  const int lane = (int)(threadIdx.x & 31u);
  const int r    = lane & 15;
  const int kh   = (lane >> 4) << 3;          // 0 or 8
  const bf16* p  = base + (size_t)r * stride + k0;
  v8bf lo = *(const v8bf*)(p + kh);           // K = kh .. kh+7
  v8bf hi = *(const v8bf*)(p + 16 + kh);      // K = 16+kh .. 16+kh+7
  return __builtin_shufflevector(lo, hi, 0, 1, 2, 3, 4, 5, 6, 7,
                                 8, 9, 10, 11, 12, 13, 14, 15);
}

// B fragment: 32x16 bf16 (K x N); weights stored [N][K] row-major (W used as W.T).
__device__ __forceinline__ v16bf load_b32x16(const bf16* w, int n0, int K, int k0) {
  const int lane = (int)(threadIdx.x & 31u);
  const int n    = lane & 15;
  const int kh   = (lane >> 4) << 3;          // 0 or 8
  const bf16* p  = w + (size_t)(n0 + n) * K + k0 + kh;
  v8bf lo = *(const v8bf*)(p);
  v8bf hi = *(const v8bf*)(p + 16);
  return __builtin_shufflevector(lo, hi, 0, 1, 2, 3, 4, 5, 6, 7,
                                 8, 9, 10, 11, 12, 13, 14, 15);
}

// ---------------------------------------------------------------------------
// Tensor Data Mover: 1D contiguous copy of `nelem` bf16 elements -> LDS.
// D# per CDNA5 ISA: count=1, type=2 ("image"),
// data_size=2B, single-row tile of nelem elements. Groups 2/3 unused (2D).
// This toolchain exposes the 6-arg builtin: (g0, g1, g2, g3, g4, cpol).
// ---------------------------------------------------------------------------
__device__ __forceinline__ void tdm_load_1d(unsigned lds_off,
                                            unsigned long long gaddr,
                                            unsigned nelem) {
  u32x4 g0;
  g0[0] = 1u;                                            // count=1, user mode
  g0[1] = lds_off;                                       // lds_addr  [63:32]
  g0[2] = (unsigned)(gaddr & 0xffffffffull);             // global_addr[31:0]
  g0[3] = (unsigned)((gaddr >> 32) & 0x01ffffffull)      // global_addr[56:32]
        | (2u << 30);                                    // type=2
  i32x8 g1;
  g1[0] = (int)(1u << 16);                               // data_size=1 (2 bytes)
  g1[1] = (int)((nelem & 0xffffu) << 16);                // tensor_dim0[15:0]
  g1[2] = (int)(((nelem >> 16) & 0xffffu) | (1u << 16)); // tensor_dim0[31:16], tensor_dim1=1
  g1[3] = (int)((nelem & 0xffffu) << 16);                // tile_dim0
  g1[4] = 0;                                             // tile_dim1=0, tile_dim2=0
  g1[5] = (int)nelem;                                    // tensor_dim0_stride[31:0]
  g1[6] = 0;
  g1[7] = 0;
  i32x4 z4 = {0, 0, 0, 0};
  i32x8 z8 = {0, 0, 0, 0, 0, 0, 0, 0};
  __builtin_amdgcn_tensor_load_to_lds(g0, g1, z4, z4, z8, 0);
}

// ---------------------------------------------------------------------------
// Prep kernels
// ---------------------------------------------------------------------------
__global__ void cvt_bf16_kernel(const float* __restrict__ src,
                                bf16* __restrict__ dst, int n) {
  int e = blockIdx.x * blockDim.x + threadIdx.x;
  if (e < n) dst[e] = (bf16)src[e];
}

__global__ void bn_stats_kernel(const float* __restrict__ v,
                                const int* __restrict__ lengths,
                                const float* __restrict__ g,
                                const float* __restrict__ b,
                                float* __restrict__ scale,
                                float* __restrict__ shift, int C) {
  const int c   = blockIdx.x;
  const int tid = threadIdx.x;
  float s0 = 0.f, s1 = 0.f, s2 = 0.f;   // count, sum, sumsq
  for (int idx = tid; idx < NTOK; idx += blockDim.x) {
    int srow = idx >> 9;                // / BATCH
    int bcol = idx & (BATCH - 1);
    if (srow < lengths[bcol]) {
      float val = v[(size_t)idx * C + c];
      s0 += 1.f; s1 += val; s2 += val * val;
    }
  }
  __shared__ float r0[256], r1[256], r2[256];
  r0[tid] = s0; r1[tid] = s1; r2[tid] = s2;
  __syncthreads();
  for (int off = 128; off > 0; off >>= 1) {
    if (tid < off) { r0[tid] += r0[tid+off]; r1[tid] += r1[tid+off]; r2[tid] += r2[tid+off]; }
    __syncthreads();
  }
  if (tid == 0) {
    float cnt = r0[0];
    float mu  = r1[0] / cnt;
    float var = r2[0] / cnt - mu * mu;
    float rs  = rsqrtf(var + 1e-5f);
    float sc  = g[c] * rs;
    scale[c] = sc;
    shift[c] = b[c] - mu * sc;
  }
}

__global__ void bn_apply_kernel(const float* __restrict__ src,
                                const float* __restrict__ scale,
                                const float* __restrict__ shift,
                                bf16* __restrict__ dst, int cmask, int total) {
  for (int e = blockIdx.x * blockDim.x + threadIdx.x; e < total;
       e += gridDim.x * blockDim.x) {
    int c = e & cmask;
    dst[e] = (bf16)(src[e] * scale[c] + shift[c]);
  }
}

// ---------------------------------------------------------------------------
// GRU scan: 32 blocks x 16-batch tiles, 8 waves; x_t tiles streamed by TDM
// into a double buffer. r/z pre-activations stored summed (gi+gh); n-gate
// parts kept separate (needed for  n = tanh(inn + r*hn) ).
// ---------------------------------------------------------------------------
__global__ __launch_bounds__(256) void gru_kernel(
    const bf16* __restrict__ xn, const bf16* __restrict__ Wih,
    const bf16* __restrict__ Whh, const float* __restrict__ bih,
    const float* __restrict__ bhh, const int* __restrict__ lengths,
    bf16* __restrict__ ys, float* __restrict__ hout) {
  __shared__ float sGrz[16 * 256];   // gi+gh for r,z columns (0..255)     16 KB
  __shared__ float sGni[16 * 128];   // gi for n columns                    8 KB
  __shared__ float sGnh[16 * 128];   // gh for n columns                    8 KB
  __shared__ float sHf[16 * HDIM];   //                                     8 KB
  __shared__ bf16  sHbf[16 * HDIM];  //                                     4 KB
  __shared__ bf16  sXn[2][16 * IDIM];// double-buffered x_t tile            4 KB
  __shared__ int   sLen[16];

  const int tid  = (int)threadIdx.x;
  const int lane = tid & 31;
  const int wave = tid >> 5;          // 0..7, 3 N-tiles each (8*3 = 24)
  const int bb0  = blockIdx.x * 16;
  const int mh   = (lane >> 4) * 8;
  const int nlo  = lane & 15;

  for (int i = tid; i < 16 * HDIM; i += 256) { sHf[i] = 0.f; sHbf[i] = (bf16)0.f; }
  if (tid < 16) sLen[tid] = lengths[bb0 + tid];

  float bihv[3], bhhv[3];
#pragma unroll
  for (int nt = 0; nt < 3; ++nt) {
    int col = (wave * 3 + nt) * 16 + nlo;
    bihv[nt] = bih[col];
    bhhv[nt] = bhh[col];
  }

  // kick off the async copy of step 0 (wave 0 owns the TDM pipe)
  if (wave == 0) {
    tdm_load_1d((unsigned)(size_t)&sXn[0][0],
                (unsigned long long)(size_t)(xn + (size_t)bb0 * IDIM),
                16 * IDIM);
  }

  for (int s = 0; s < S_LEN; ++s) {
    const int cur = s & 1;
    if (wave == 0) __builtin_amdgcn_s_wait_tensorcnt((short)0);
    __syncthreads();   // x_t tile + previous h update visible to all waves
    if (wave == 0 && s + 1 < S_LEN) {
      tdm_load_1d((unsigned)(size_t)&sXn[cur ^ 1][0],
                  (unsigned long long)(size_t)(xn + ((size_t)(s + 1) * BATCH + bb0) * IDIM),
                  16 * IDIM);
    }

    // A operands: h (LDS bf16) and x_t (LDS bf16, TDM-staged)
    v16bf ah[4];
#pragma unroll
    for (int kt = 0; kt < 4; ++kt) ah[kt] = load_a16x32(sHbf, HDIM, kt * 32);
    v16bf ax[2];
#pragma unroll
    for (int kt = 0; kt < 2; ++kt) ax[kt] = load_a16x32(sXn[cur], IDIM, kt * 32);

#pragma unroll
    for (int nt = 0; nt < 3; ++nt) {
      const int n0 = (wave * 3 + nt) * 16;
      v8f accg = {0.f, 0.f, 0.f, 0.f, 0.f, 0.f, 0.f, 0.f};
#pragma unroll
      for (int kt = 0; kt < 4; ++kt)
        accg = wmma_bf(ah[kt], load_b32x16(Whh, n0, HDIM, kt * 32), accg);
      v8f acci = {0.f, 0.f, 0.f, 0.f, 0.f, 0.f, 0.f, 0.f};
#pragma unroll
      for (int kt = 0; kt < 2; ++kt)
        acci = wmma_bf(ax[kt], load_b32x16(Wih, n0, IDIM, kt * 32), acci);
      const int col = n0 + nlo;
      if (col < 256) {                // r,z region: only the sum is needed
#pragma unroll
        for (int r = 0; r < 8; ++r)
          sGrz[(mh + r) * 256 + col] = accg[r] + acci[r] + bihv[nt] + bhhv[nt];
      } else {                        // n region: keep gi / gh separate
        const int cj = col - 256;
#pragma unroll
        for (int r = 0; r < 8; ++r) {
          sGni[(mh + r) * 128 + cj] = acci[r] + bihv[nt];
          sGnh[(mh + r) * 128 + cj] = accg[r] + bhhv[nt];
        }
      }
    }
    __syncthreads();

    // gate math: 16x128 states / 256 threads (PyTorch r,z,n order)
#pragma unroll
    for (int i = 0; i < 8; ++i) {
      int e = tid + i * 256;
      int brow = e >> 7, j = e & 127;
      float xr = sGrz[brow * 256 + j];
      float xz = sGrz[brow * 256 + 128 + j];
      float rg = 1.f / (1.f + __expf(-xr));
      float zg = 1.f / (1.f + __expf(-xz));
      float ng = tanhf(sGni[brow * 128 + j] + rg * sGnh[brow * 128 + j]);
      float hprev = sHf[brow * HDIM + j];
      float hnew  = (1.f - zg) * ng + zg * hprev;
      bool  valid = s < sLen[brow];
      float hcar  = valid ? hnew : hprev;
      sHf[brow * HDIM + j]  = hcar;
      sHbf[brow * HDIM + j] = (bf16)hcar;
      ys[((size_t)s * BATCH + bb0 + brow) * HDIM + j] =
          valid ? (bf16)hnew : (bf16)0.f;
    }
    __syncthreads();
  }

#pragma unroll
  for (int i = 0; i < 8; ++i) {
    int e = tid + i * 256;
    int brow = e >> 7, j = e & 127;
    hout[(size_t)(bb0 + brow) * HDIM + j] = sHf[brow * HDIM + j];
  }
}

// ---------------------------------------------------------------------------
// MLP: 16-token tiles, relu(W1@[y,xdn]) -> relu(W2@) -> relu(W3@) -> mask
// ---------------------------------------------------------------------------
__global__ __launch_bounds__(128) void mlp_kernel(
    const bf16* __restrict__ ysb, const bf16* __restrict__ xdn,
    const bf16* __restrict__ W1, const float* __restrict__ b1,
    const bf16* __restrict__ W2, const float* __restrict__ b2,
    const bf16* __restrict__ W3, const float* __restrict__ b3,
    const int* __restrict__ lengths, float* __restrict__ out) {
  __shared__ bf16 sD1[16 * HDIM];
  __shared__ bf16 sD2[16 * HDIM];
  const int tid  = (int)threadIdx.x;
  const int lane = tid & 31;
  const int wave = tid >> 5;
  const int t0   = blockIdx.x * 16;
  const int mh   = (lane >> 4) * 8;
  const int nlo  = lane & 15;

  // ---- layer 1: A = [ys(128) | xdn(32)] ----
  v16bf ay[4];
  const bf16* ybase = ysb + (size_t)t0 * HDIM;
#pragma unroll
  for (int kt = 0; kt < 4; ++kt) ay[kt] = load_a16x32(ybase, HDIM, kt * 32);
  v16bf axd = load_a16x32(xdn + (size_t)t0 * DDIM, DDIM, 0);

#pragma unroll
  for (int nt = 0; nt < 2; ++nt) {
    int n0 = (wave * 2 + nt) * 16;
    v8f acc = {0.f, 0.f, 0.f, 0.f, 0.f, 0.f, 0.f, 0.f};
#pragma unroll
    for (int kt = 0; kt < 4; ++kt)
      acc = wmma_bf(ay[kt], load_b32x16(W1, n0, HDIM + DDIM, kt * 32), acc);
    acc = wmma_bf(axd, load_b32x16(W1, n0, HDIM + DDIM, HDIM), acc);
    int col = n0 + nlo;
    float bias = b1[col];
#pragma unroll
    for (int r = 0; r < 8; ++r)
      sD1[(mh + r) * HDIM + col] = (bf16)fmaxf(acc[r] + bias, 0.f);
  }
  __syncthreads();

  // ---- layer 2 ----
  v16bf ad[4];
#pragma unroll
  for (int kt = 0; kt < 4; ++kt) ad[kt] = load_a16x32(sD1, HDIM, kt * 32);
#pragma unroll
  for (int nt = 0; nt < 2; ++nt) {
    int n0 = (wave * 2 + nt) * 16;
    v8f acc = {0.f, 0.f, 0.f, 0.f, 0.f, 0.f, 0.f, 0.f};
#pragma unroll
    for (int kt = 0; kt < 4; ++kt)
      acc = wmma_bf(ad[kt], load_b32x16(W2, n0, HDIM, kt * 32), acc);
    int col = n0 + nlo;
    float bias = b2[col];
#pragma unroll
    for (int r = 0; r < 8; ++r)
      sD2[(mh + r) * HDIM + col] = (bf16)fmaxf(acc[r] + bias, 0.f);
  }
  __syncthreads();

  // ---- layer 3 + masked f32 store ----
  v16bf ad2[4];
#pragma unroll
  for (int kt = 0; kt < 4; ++kt) ad2[kt] = load_a16x32(sD2, HDIM, kt * 32);
#pragma unroll
  for (int nt = 0; nt < 2; ++nt) {
    int n0 = (wave * 2 + nt) * 16;
    v8f acc = {0.f, 0.f, 0.f, 0.f, 0.f, 0.f, 0.f, 0.f};
#pragma unroll
    for (int kt = 0; kt < 4; ++kt)
      acc = wmma_bf(ad2[kt], load_b32x16(W3, n0, HDIM, kt * 32), acc);
    int col = n0 + nlo;
    float bias = b3[col];
#pragma unroll
    for (int r = 0; r < 8; ++r) {
      int token = t0 + mh + r;
      int srow  = token >> 9;
      int bcol  = token & (BATCH - 1);
      float v = fmaxf(acc[r] + bias, 0.f);
      out[(size_t)token * HDIM + col] = (srow < lengths[bcol]) ? v : 0.f;
    }
  }
}

// ---------------------------------------------------------------------------
extern "C" void kernel_launch(void* const* d_in, const int* in_sizes, int n_in,
                              void* d_out, int out_size, void* d_ws, size_t ws_size,
                              hipStream_t stream) {
  (void)in_sizes; (void)n_in; (void)out_size; (void)ws_size;

  const float* x     = (const float*)d_in[0];
  const float* xd    = (const float*)d_in[1];
  const int*   lens  = (const int*)d_in[2];
  const float* bn_g  = (const float*)d_in[3];
  const float* bn_b  = (const float*)d_in[4];
  const float* bnd_g = (const float*)d_in[5];
  const float* bnd_b = (const float*)d_in[6];
  const float* Wih   = (const float*)d_in[7];
  const float* Whh   = (const float*)d_in[8];
  const float* bih   = (const float*)d_in[9];
  const float* bhh   = (const float*)d_in[10];
  // d_in[11], d_in[12]: attn_W / attn_b -- provably identity in the reference.
  const float* W1 = (const float*)d_in[13];
  const float* b1 = (const float*)d_in[14];
  const float* W2 = (const float*)d_in[15];
  const float* b2 = (const float*)d_in[16];
  const float* W3 = (const float*)d_in[17];
  const float* b3 = (const float*)d_in[18];

  char* ws = (char*)d_ws;
  float* scaleI = (float*)(ws + 0);
  float* shiftI = (float*)(ws + 256);
  float* scaleD = (float*)(ws + 512);
  float* shiftD = (float*)(ws + 768);
  bf16* WihB = (bf16*)(ws + 1024);                       // 384*64*2  = 49152
  bf16* WhhB = (bf16*)(ws + 50176);                      // 384*128*2 = 98304
  bf16* W1B  = (bf16*)(ws + 148480);                     // 128*160*2 = 40960
  bf16* W2B  = (bf16*)(ws + 189440);                     // 128*128*2 = 32768
  bf16* W3B  = (bf16*)(ws + 222208);                     // 128*128*2 = 32768
  bf16* xnB  = (bf16*)(ws + 254976);                     // NTOK*64*2  = 16 MiB
  bf16* xdnB = (bf16*)(ws + 254976 + 16777216);          // NTOK*32*2  = 8 MiB
  bf16* ysB  = (bf16*)(ws + 254976 + 16777216 + 8388608);// NTOK*128*2 = 32 MiB

  float* outp  = (float*)d_out;
  float* hlast = outp + (size_t)S_LEN * BATCH * HDIM;

  // weight f32 -> bf16
  cvt_bf16_kernel<<<(384 * 64 + 255) / 256, 256, 0, stream>>>(Wih, WihB, 384 * 64);
  cvt_bf16_kernel<<<(384 * 128 + 255) / 256, 256, 0, stream>>>(Whh, WhhB, 384 * 128);
  cvt_bf16_kernel<<<(128 * 160 + 255) / 256, 256, 0, stream>>>(W1, W1B, 128 * 160);
  cvt_bf16_kernel<<<(128 * 128 + 255) / 256, 256, 0, stream>>>(W2, W2B, 128 * 128);
  cvt_bf16_kernel<<<(128 * 128 + 255) / 256, 256, 0, stream>>>(W3, W3B, 128 * 128);

  // masked batchnorm
  bn_stats_kernel<<<IDIM, 256, 0, stream>>>(x, lens, bn_g, bn_b, scaleI, shiftI, IDIM);
  bn_stats_kernel<<<DDIM, 256, 0, stream>>>(xd, lens, bnd_g, bnd_b, scaleD, shiftD, DDIM);
  bn_apply_kernel<<<2048, 256, 0, stream>>>(x, scaleI, shiftI, xnB, IDIM - 1, NTOK * IDIM);
  bn_apply_kernel<<<2048, 256, 0, stream>>>(xd, scaleD, shiftD, xdnB, DDIM - 1, NTOK * DDIM);

  // GRU scan (TDM-streamed inputs; writes ys bf16 + final hidden into d_out tail)
  gru_kernel<<<BATCH / 16, 256, 0, stream>>>(xnB, WihB, WhhB, bih, bhh, lens, ysB, hlast);

  // fused 3-layer MLP + length mask
  mlp_kernel<<<NTOK / 16, 128, 0, stream>>>(ysB, xdnB, W1B, b1, W2B, b2, W3B, b3,
                                            lens, outp);
}
